// BasicMTL_49014166781960
// MI455X (gfx1250) — matrix operand order repaired and checked by reading, not compiled, not run on hardware
//
#include <hip/hip_runtime.h>
#include <stdint.h>

typedef __bf16 bf16;
typedef __attribute__((ext_vector_type(16))) __bf16 v16bf;
typedef __attribute__((ext_vector_type(8)))  __bf16 v8bf;
typedef __attribute__((ext_vector_type(8)))  float  v8f;

#define T_TOK 4096
#define NC    4096
#define H     1024
#define META  20
#define KD    3092   // 3*H + META
#define KP    3104   // K padded to multiple of 32 (bf16 WMMA K-step)
#define NP2   3136   // GEMM2 N padded to multiple of 64 (wave macro-tile width)
#define UH    1024
#define MTOP  1024
#define KANT  50
#define NEGF  (-1e30f)

__device__ __forceinline__ unsigned short f2bf(float f) {
  unsigned u = __builtin_bit_cast(unsigned, f);
  u += 0x7FFFu + ((u >> 16) & 1u);          // round-to-nearest-even
  return (unsigned short)(u >> 16);
}

// ---------------- token logits: tl[t] = hidden[t,:] . w_attn + b ----------------
__global__ void k_token_logits(const float* __restrict__ hid, const float* __restrict__ w_attn,
                               const float* __restrict__ b_attn, float* __restrict__ tl) {
  __shared__ float red[256];
  int t = blockIdx.x, tid = threadIdx.x;
  const float* row = hid + (size_t)t * H;
  float s = 0.f;
  for (int h = tid; h < H; h += 256) s += row[h] * w_attn[h];
  red[tid] = s; __syncthreads();
  for (int off = 128; off > 0; off >>= 1) { if (tid < off) red[tid] += red[tid + off]; __syncthreads(); }
  if (tid == 0) tl[t] = red[0] + b_attn[0];
}

// ---------------- span embeddings (bf16, K padded to 3104) ----------------
__global__ void k_span_emb(const float* __restrict__ hid, const int* __restrict__ starts,
                           const int* __restrict__ widths, const float* __restrict__ w_width,
                           const float* __restrict__ tl, unsigned short* __restrict__ se) {
  __shared__ float p[1024];
  __shared__ float red[256];
  __shared__ float s_m, s_z;
  int c = blockIdx.x, tid = threadIdx.x;
  int s = starts[c], w = widths[c], e = s + w, len = w + 1;
  float lm = -3.4e38f;
  for (int t = tid; t < len; t += 256) lm = fmaxf(lm, tl[s + t]);
  red[tid] = lm; __syncthreads();
  for (int off = 128; off > 0; off >>= 1) { if (tid < off) red[tid] = fmaxf(red[tid], red[tid + off]); __syncthreads(); }
  if (tid == 0) s_m = red[0];
  __syncthreads();
  float m = s_m, ls = 0.f;
  for (int t = tid; t < len; t += 256) { float ex = __expf(tl[s + t] - m); p[t] = ex; ls += ex; }
  red[tid] = ls; __syncthreads();
  for (int off = 128; off > 0; off >>= 1) { if (tid < off) red[tid] += red[tid + off]; __syncthreads(); }
  if (tid == 0) s_z = red[0];
  __syncthreads();
  float inv = 1.f / s_z;
  float acc[4] = {0.f, 0.f, 0.f, 0.f};
  for (int t = 0; t < len; ++t) {            // L2-resident attention pooling
    float wt = p[t] * inv;
    const float* hp = hid + (size_t)(s + t) * H;
#pragma unroll
    for (int r = 0; r < 4; ++r) acc[r] += wt * hp[tid + 256 * r];
  }
  unsigned short* row = se + (size_t)c * KP;
  const float* hs = hid + (size_t)s * H;
  const float* he = hid + (size_t)e * H;
#pragma unroll
  for (int r = 0; r < 4; ++r) {
    int h = tid + 256 * r;
    row[h]                = f2bf(hs[h]);
    row[H + h]            = f2bf(he[h]);
    row[2 * H + META + h] = f2bf(acc[r]);
  }
  if (tid < META) row[2 * H + tid] = f2bf(w_width[(size_t)w * META + tid]);
  if (tid < KP - KD) row[KD + tid] = 0;
}

// ---------------- weight converts: B stored column-major bf16 [N][KP] ----------------
__global__ void k_conv_wu1(const float* __restrict__ w_u1, unsigned short* __restrict__ bt) {
  int n = blockIdx.x, tid = threadIdx.x;
  unsigned short* dst = bt + (size_t)n * KP;
  for (int k = tid; k < KP; k += 256)
    dst[k] = (k < KD) ? f2bf(w_u1[(size_t)k * UH + n]) : (unsigned short)0;
}

__global__ void k_conv_wfast(const float* __restrict__ w_fast, const float* __restrict__ b_fast,
                             unsigned short* __restrict__ bt, float* __restrict__ bias_pad) {
  int n = blockIdx.x, tid = threadIdx.x;   // n in [0, NP2)
  unsigned short* dst = bt + (size_t)n * KP;
  bool nv = n < KD;
  for (int k = tid; k < KP; k += 256)
    dst[k] = (nv && k < KD) ? f2bf(w_fast[(size_t)k * KD + n]) : (unsigned short)0;
  if (tid == 0) bias_pad[n] = nv ? b_fast[n] : 0.f;
}

// ---------------- bf16 WMMA GEMM with 32x64 register blocking per wave ----------------
// C = A[M,K] * Bt[N,K]^T (+bias, opt relu). One wave owns a 32x64 macro-tile:
// 2 M-tiles x 4 N-tiles = 8 accumulators; per 32-deep K-step: 12 global_load_b128,
// 8 v_wmma_f32_16x16x32_bf16 (A frags reused 4x, B frags reused 2x).
// __launch_bounds__(256, 1): lift the default VGPR cap so the software-pipelined
// loop keeps all fragments in registers (no scratch spills).
__global__ void __launch_bounds__(256, 1)
k_wmma_gemm(const unsigned short* __restrict__ A,
            const unsigned short* __restrict__ Bt,
            const float* __restrict__ bias,
            float* __restrict__ C, unsigned short* __restrict__ Cb,
            int M, int N, int K, int lda, int ldb, int ldc, int relu) {
  int wid  = (int)((blockIdx.x * blockDim.x + threadIdx.x) >> 5);
  int lane = (int)(threadIdx.x & 31);
  int nblocks = N >> 6;                        // 64-wide N blocks
  int mb = wid / nblocks, nb = wid - mb * nblocks;
  if (mb * 32 >= M) return;
  bool hi = lane >= 16;
  int aoff = hi ? 8 : 0;                       // A half-K offset per lane group
  int boff = hi ? 8 : 0;                       // B half-K offset per lane group
  const bf16* Ar[2];
  const bf16* Bc[4];
#pragma unroll
  for (int i = 0; i < 2; ++i)
    Ar[i] = (const bf16*)A + (size_t)(mb * 32 + i * 16 + (lane & 15)) * lda;
#pragma unroll
  for (int j = 0; j < 4; ++j)
    Bc[j] = (const bf16*)Bt + (size_t)(nb * 64 + j * 16 + (lane & 15)) * ldb;

  v8f acc[2][4];
#pragma unroll
  for (int i = 0; i < 2; ++i)
#pragma unroll
    for (int j = 0; j < 4; ++j) acc[i][j] = (v8f){0.f,0.f,0.f,0.f,0.f,0.f,0.f,0.f};

  for (int k0 = 0; k0 < K; k0 += 32) {
    v16bf av[2], bv[4];
#pragma unroll
    for (int i = 0; i < 2; ++i) {
      v8bf lo = *reinterpret_cast<const v8bf*>(Ar[i] + k0 + aoff);
      v8bf hi8 = *reinterpret_cast<const v8bf*>(Ar[i] + k0 + 16 + aoff);
#pragma unroll
      for (int q = 0; q < 8; ++q) { av[i][q] = lo[q]; av[i][q + 8] = hi8[q]; }
    }
#pragma unroll
    for (int j = 0; j < 4; ++j) {
      v8bf lo = *reinterpret_cast<const v8bf*>(Bc[j] + k0 + boff);
      v8bf hi8 = *reinterpret_cast<const v8bf*>(Bc[j] + k0 + 16 + boff);
#pragma unroll
      for (int q = 0; q < 8; ++q) { bv[j][q] = lo[q]; bv[j][q + 8] = hi8[q]; }
    }
    if (k0 + 64 < K) {                         // K-tile prefetch two steps ahead (global_prefetch_b8)
      __builtin_prefetch(Ar[0] + k0 + 64, 0, 1);
      __builtin_prefetch(Bc[0] + k0 + 64, 0, 1);
      __builtin_prefetch(Bc[2] + k0 + 64, 0, 1);
    }
#pragma unroll
    for (int i = 0; i < 2; ++i)
#pragma unroll
      for (int j = 0; j < 4; ++j)
        acc[i][j] = __builtin_amdgcn_wmma_f32_16x16x32_bf16(
            false, av[i], false, bv[j], (short)0, acc[i][j], false, false);
  }
#pragma unroll
  for (int i = 0; i < 2; ++i) {
#pragma unroll
    for (int j = 0; j < 4; ++j) {
      int n = nb * 64 + j * 16 + (lane & 15);
      float bval = bias ? bias[n] : 0.f;
#pragma unroll
      for (int r = 0; r < 8; ++r) {
        int m = mb * 32 + i * 16 + r + (hi ? 8 : 0);
        float v = acc[i][j][r] + bval;
        if (relu) v = fmaxf(v, 0.f);
        if (C)  C[(size_t)m * ldc + n]  = v;
        if (Cb) Cb[(size_t)m * ldc + n] = f2bf(v);
      }
    }
  }
}

// ---------------- mention scores: relu_out . w_u2 + b_u2 ----------------
__global__ void k_mention(const float* __restrict__ ro, const float* __restrict__ w_u2,
                          const float* __restrict__ b_u2, float* __restrict__ msc) {
  __shared__ float red[256];
  int c = blockIdx.x, tid = threadIdx.x;
  const float* row = ro + (size_t)c * UH;
  float s = 0.f;
  for (int h = tid; h < UH; h += 256) s += row[h] * w_u2[h];
  red[tid] = s; __syncthreads();
  for (int off = 128; off > 0; off >>= 1) { if (tid < off) red[tid] += red[tid + off]; __syncthreads(); }
  if (tid == 0) msc[c] = red[0] + b_u2[0];
}

// ---------------- greedy non-crossing span selection (single workgroup) ----------------
__global__ void __launch_bounds__(1024) k_extract(const int* __restrict__ starts,
                                                  const int* __restrict__ widths,
                                                  const float* __restrict__ msc,
                                                  int* __restrict__ flags, int* __restrict__ topidx) {
  __shared__ unsigned long long key[NC];   // 32 KB, later reused as ord[]/earliest_start[]
  __shared__ int s_le[T_TOK];              // latest_end, 16 KB
  __shared__ int s_flag;
  int tid = threadIdx.x;
  for (int i = tid; i < NC; i += 1024) {
    unsigned u = __builtin_bit_cast(unsigned, msc[i]);
    u = (u & 0x80000000u) ? ~u : (u | 0x80000000u);
    key[i] = ((unsigned long long)u << 32) | (unsigned long long)(0xFFFFFFFFu - (unsigned)i);
  }
  __syncthreads();
  for (int size = 2; size <= NC; size <<= 1) {
    for (int stride = size >> 1; stride > 0; stride >>= 1) {
      for (int t = tid; t < NC / 2; t += 1024) {
        int lo = ((t / stride) * (stride << 1)) + (t % stride);
        int h2 = lo + stride;
        bool dir = ((lo & size) == 0);
        unsigned long long a = key[lo], b = key[h2];
        if (dir ? (a < b) : (a > b)) { key[lo] = b; key[h2] = a; }
      }
      __syncthreads();
    }
  }
  int* s_ord = (int*)key;
  int* s_es  = ((int*)key) + NC;
  int myord[4];
#pragma unroll
  for (int j = 0; j < 4; ++j)
    myord[j] = (int)(0xFFFFFFFFu - (unsigned)(key[tid + j * 1024] & 0xFFFFFFFFull));
  __syncthreads();
#pragma unroll
  for (int j = 0; j < 4; ++j) {
    int i = tid + j * 1024;
    s_ord[i] = myord[j];
    s_es[i]  = T_TOK;
    s_le[i]  = -1;
  }
  __syncthreads();
  int count = 0;
  for (int it = 0; it < NC; ++it) {
    int cand = s_ord[it];
    int si = starts[cand];
    int ei = si + widths[cand];
    if (tid == 0) s_flag = 0;
    __syncthreads();
    for (int t = si + tid; t <= ei; t += 1024) {
      if (t > si && s_le[t] > ei) s_flag = 1;
      if (t < ei && s_es[t] < si) s_flag = 1;
    }
    __syncthreads();
    if (tid == 0) {
      int acc = (s_flag == 0) && (count < MTOP);
      if (acc) {
        if (ei > s_le[si]) s_le[si] = ei;
        if (si < s_es[ei]) s_es[ei] = si;
        count++;
      }
      flags[cand] = acc;
    }
    __syncthreads();
  }
  if (tid == 0) {
    int n = 0;
    for (int c2 = 0; c2 < NC && n < MTOP; ++c2) if (flags[c2]) topidx[n++] = c2;
    while (n < MTOP) topidx[n++] = NC - 1;
  }
}

// ---------------- gather top mentions ----------------
__global__ void k_gather(const unsigned short* __restrict__ se, const float* __restrict__ msc,
                         const int* __restrict__ topidx, unsigned short* __restrict__ temb,
                         float* __restrict__ topm, int* __restrict__ out_idx) {
  int m = blockIdx.x, tid = threadIdx.x;
  int idx = topidx[m];
  const unsigned short* src = se + (size_t)idx * KP;
  unsigned short* dst = temb + (size_t)m * KP;
  for (int k = tid; k < KP; k += 256) dst[k] = src[k];
  if (tid == 0) { topm[m] = msc[idx]; out_idx[m] = idx; }
}

// ---------------- fast scores + per-row top-k(50) ----------------
__global__ void k_topk(const float* __restrict__ pair, const float* __restrict__ topm,
                       int* __restrict__ out_ant, float* __restrict__ out_mask,
                       float* __restrict__ out_score) {
  __shared__ float vals[MTOP];
  __shared__ float rb[256];
  __shared__ int   ri[256];
  int i = blockIdx.x, tid = threadIdx.x;
  float tmi = topm[i];
  for (int j = tid; j < MTOP; j += 256) {
    float v = pair[(size_t)i * MTOP + j] + tmi + topm[j];
    if (i - j < 1) v += NEGF;
    vals[j] = v;
  }
  __syncthreads();
  for (int k = 0; k < KANT; ++k) {
    float bv = -3.4e38f; int bj = 0;
    for (int j = tid; j < MTOP; j += 256) {
      float v = vals[j];
      if (v > bv || (v == bv && j < bj)) { bv = v; bj = j; }
    }
    rb[tid] = bv; ri[tid] = bj; __syncthreads();
    for (int off = 128; off > 0; off >>= 1) {
      if (tid < off) {
        float ov = rb[tid + off]; int oj = ri[tid + off];
        if (ov > rb[tid] || (ov == rb[tid] && oj < ri[tid])) { rb[tid] = ov; ri[tid] = oj; }
      }
      __syncthreads();
    }
    if (tid == 0) {
      int j = ri[0];
      out_ant[(size_t)i * KANT + k]   = j;
      out_mask[(size_t)i * KANT + k]  = (i - j >= 1) ? 1.0f : 0.0f;
      out_score[(size_t)i * KANT + k] = rb[0];
      vals[j] = -3.4e38f;
    }
    __syncthreads();
  }
}

extern "C" void kernel_launch(void* const* d_in, const int* in_sizes, int n_in,
                              void* d_out, int out_size, void* d_ws, size_t ws_size,
                              hipStream_t stream) {
  (void)in_sizes; (void)n_in; (void)out_size; (void)ws_size;
  const float* hid     = (const float*)d_in[0];
  const int*   starts  = (const int*)d_in[1];
  const int*   widths  = (const int*)d_in[2];
  const float* w_width = (const float*)d_in[5];
  const float* w_attn  = (const float*)d_in[6];
  const float* b_attn  = (const float*)d_in[7];
  const float* w_u1    = (const float*)d_in[8];
  const float* b_u1    = (const float*)d_in[9];
  const float* w_u2    = (const float*)d_in[10];
  const float* b_u2    = (const float*)d_in[11];
  const float* w_fast  = (const float*)d_in[12];
  const float* b_fast  = (const float*)d_in[13];

  char* ws = (char*)d_ws;
  size_t off = 0;
  auto take = [&](size_t bytes) -> char* {
    char* p = ws + off; off += bytes; off = (off + 255) & ~(size_t)255; return p;
  };
  float*          tl     = (float*)take((size_t)T_TOK * 4);
  unsigned short* se     = (unsigned short*)take((size_t)NC * KP * 2);
  unsigned short* wu1t   = (unsigned short*)take((size_t)UH * KP * 2);
  float*          relu_o = (float*)take((size_t)NC * UH * 4);
  float*          msc    = (float*)take((size_t)NC * 4);
  int*            flags  = (int*)take((size_t)NC * 4);
  int*            topidx = (int*)take((size_t)MTOP * 4);
  unsigned short* temb   = (unsigned short*)take((size_t)MTOP * KP * 2);
  float*          topm   = (float*)take((size_t)MTOP * 4);
  unsigned short* wft    = (unsigned short*)take((size_t)NP2 * KP * 2);
  float*          bfp    = (float*)take((size_t)NP2 * 4);
  unsigned short* tmpb   = (unsigned short*)take((size_t)MTOP * NP2 * 2);
  float*          pairm  = (float*)take((size_t)MTOP * MTOP * 4);

  int*   out_idx   = (int*)d_out;                  // [1024] (int bits)
  int*   out_ant   = out_idx + MTOP;               // [1024*50]
  float* out_mask  = (float*)(out_ant + MTOP * KANT);
  float* out_score = out_mask + MTOP * KANT;

  k_token_logits<<<T_TOK, 256, 0, stream>>>(hid, w_attn, b_attn, tl);
  k_conv_wu1<<<UH, 256, 0, stream>>>(w_u1, wu1t);
  k_conv_wfast<<<NP2, 256, 0, stream>>>(w_fast, b_fast, wft, bfp);
  k_span_emb<<<NC, 256, 0, stream>>>(hid, starts, widths, w_width, tl, se);
  // GEMM1: relu(span_emb @ w_u1 + b_u1)  [4096 x 1024], waves = (4096/32)*(1024/64) = 2048
  k_wmma_gemm<<<(NC / 32) * (UH / 64) / 8, 256, 0, stream>>>(
      se, wu1t, b_u1, relu_o, nullptr, NC, UH, KP, KP, KP, UH, 1);
  k_mention<<<NC, 256, 0, stream>>>(relu_o, w_u2, b_u2, msc);
  k_extract<<<1, 1024, 0, stream>>>(starts, widths, msc, flags, topidx);
  k_gather<<<MTOP, 256, 0, stream>>>(se, msc, topidx, temb, topm, out_idx);
  // GEMM2: tmp = top_emb @ w_fast + b_fast  [1024 x 3136pad] bf16 out, waves = 32*49 = 1568
  k_wmma_gemm<<<(MTOP / 32) * (NP2 / 64) / 8, 256, 0, stream>>>(
      temb, wft, bfp, nullptr, tmpb, MTOP, NP2, KP, KP, KP, NP2, 0);
  // GEMM3: pair = tmp @ top_emb^T  [1024 x 1024], waves = 32*16 = 512
  k_wmma_gemm<<<(MTOP / 32) * (MTOP / 64) / 8, 256, 0, stream>>>(
      tmpb, temb, nullptr, pairm, nullptr, MTOP, MTOP, KP, NP2, KP, MTOP, 0);
  k_topk<<<MTOP, 256, 0, stream>>>(pairm, topm, out_ant, out_mask, out_score);
}
